// MambaTower_9337258902063
// MI455X (gfx1250) — matrix operand order repaired and checked by reading, not compiled, not run on hardware
//
#include <hip/hip_runtime.h>
#include <hip/hip_bf16.h>
#include <cstdint>

// ---------------------------------------------------------------------------
// Mamba tower (4 layers) for gfx1250:
//   - f16 WMMA GEMMs (f32 accum), weights pre-converted to f16
//   - B operand staged via GLOBAL_LOAD_ASYNC_TO_LDS_B128 (ASYNCcnt pipelined)
//   - double-buffered LDS, 2x K-step software pipeline (8 static v_wmma)
//   - fp32 selective scan, conv+SiLU, residual+LayerNorm
// ---------------------------------------------------------------------------

typedef __attribute__((ext_vector_type(16))) _Float16 v16h;
typedef __attribute__((ext_vector_type(8)))  _Float16 v8h;
typedef __attribute__((ext_vector_type(8)))  float    v8f;

#define D_MODEL 1024
#define D_INNER 2048
#define D_STATE 16
#define D_CONV  4
#define DT_RANK 64
#define BATCH   2
#define SEQ     1024
#define TOKENS  (BATCH * SEQ)   // 2048
#define LN_EPS  1e-5f

#define TILE_M 64
#define TILE_N 128
#define TILE_K 32

__device__ inline v8h cvt8_f32_to_f16(const float* __restrict__ src) {
  float4 f0 = *(const float4*)(src);
  float4 f1 = *(const float4*)(src + 4);
  v8h h;
  h[0] = (_Float16)f0.x; h[1] = (_Float16)f0.y;
  h[2] = (_Float16)f0.z; h[3] = (_Float16)f0.w;
  h[4] = (_Float16)f1.x; h[5] = (_Float16)f1.y;
  h[6] = (_Float16)f1.z; h[7] = (_Float16)f1.w;
  return h;
}

// fp32 -> f16 bulk converter (8 elements / thread)
__global__ __launch_bounds__(256) void cvt_w_kernel(
    const float* __restrict__ src, _Float16* __restrict__ dst, int n8)
{
  const int i = blockIdx.x * 256 + threadIdx.x;
  if (i < n8) {
    *(v8h*)(dst + (size_t)i * 8) = cvt8_f32_to_f16(src + (size_t)i * 8);
  }
}

// C[M,N] = epilogue( A[M,K] (fp32, lda) * Wh[N,K]^T (f16, ldw) )
// epilogue 0: identity;  1: softplus(acc + bias[n])
// Requires: M % 64 == 0, K % 64 == 0; N guarded.
__global__ __launch_bounds__(256) void gemm_wmma_async_kernel(
    const float* __restrict__ A, int lda,
    const _Float16* __restrict__ Wh, int ldw,
    float* __restrict__ C, int ldc,
    int N, int K,
    const float* __restrict__ bias, int epilogue)
{
  __shared__ __align__(32) _Float16 As[2][TILE_M][TILE_K];  // 2 x 4 KB
  __shared__ __align__(32) _Float16 Bs[2][TILE_N][TILE_K];  // 2 x 8 KB

  const int tid   = threadIdx.x;
  const int lane  = tid & 31;            // wave32
  const int wave  = tid >> 5;            // 8 waves
  const int wm    = (wave >> 2) * 32;    // wave M offset in tile
  const int wn    = (wave &  3) * 32;    // wave N offset in tile
  const int tileM = blockIdx.y * TILE_M;
  const int tileN = blockIdx.x * TILE_N;

  v8f acc00 = {}; v8f acc01 = {}; v8f acc10 = {}; v8f acc11 = {};

  // global->LDS fill mapping
  const int arow = tid >> 2;             // 0..63
  const int acol = (tid & 3) * 8;        // 0,8,16,24
  const int brow = tid >> 1;             // 0..127
  const int bcol = (tid & 1) * 16;       // 0,16

  const int  gn     = tileN + brow;
  const bool bvalid = (gn < N);
  const _Float16* bsrc = Wh + (size_t)gn * ldw + bcol;
  const float*    asrc = A + (size_t)(tileM + arow) * lda + acol;

  // low 32 bits of a flat LDS pointer == LDS byte offset
  const unsigned bdst0 = (unsigned)(uintptr_t)(void*)&Bs[0][brow][bcol];
  const unsigned bdst1 = (unsigned)(uintptr_t)(void*)&Bs[1][brow][bcol];

  // async copy of 32 bytes (16 halves) of one weight row into LDS
  auto issueB = [&](int kb, unsigned dst) {
    if (bvalid) {
      const uint64_t g = (uint64_t)(uintptr_t)(bsrc + kb);
      asm volatile("global_load_async_to_lds_b128 %0, %1, off"
                   :: "v"(dst), "v"(g) : "memory");
      asm volatile("global_load_async_to_lds_b128 %0, %1, off offset:16"
                   :: "v"(dst), "v"(g) : "memory");
    }
  };
  auto waitAsync = [&]() {
    asm volatile("s_wait_asynccnt 0x0" ::: "memory");
  };

  const int laneLow = (lane < 16) ? 1 : 0;
  const int mrow    = lane & 15;

  auto compute = [&](int buf) {
    v16h a0, a1, b0, b1;
    {
      const int k0 = laneLow ? 0 : 16;
      v8h lo = *(const v8h*)&As[buf][wm + mrow][k0];
      v8h hi = *(const v8h*)&As[buf][wm + mrow][k0 + 8];
      a0 = __builtin_shufflevector(lo, hi, 0,1,2,3,4,5,6,7,8,9,10,11,12,13,14,15);
      lo = *(const v8h*)&As[buf][wm + 16 + mrow][k0];
      hi = *(const v8h*)&As[buf][wm + 16 + mrow][k0 + 8];
      a1 = __builtin_shufflevector(lo, hi, 0,1,2,3,4,5,6,7,8,9,10,11,12,13,14,15);
    }
    {
      const int k0 = laneLow ? 0 : 16;
      b0 = *(const v16h*)&Bs[buf][wn + mrow][k0];
      b1 = *(const v16h*)&Bs[buf][wn + 16 + mrow][k0];
    }
    acc00 = __builtin_amdgcn_wmma_f32_16x16x32_f16(false, a0, false, b0, (short)0, acc00, false, false);
    acc01 = __builtin_amdgcn_wmma_f32_16x16x32_f16(false, a0, false, b1, (short)0, acc01, false, false);
    acc10 = __builtin_amdgcn_wmma_f32_16x16x32_f16(false, a1, false, b0, (short)0, acc10, false, false);
    acc11 = __builtin_amdgcn_wmma_f32_16x16x32_f16(false, a1, false, b1, (short)0, acc11, false, false);
  };

  // ---- prologue: fill stage 0 ----
  issueB(0, bdst0);
  v8h aR = cvt8_f32_to_f16(asrc);
  *(v8h*)&As[0][arow][acol] = aR;
  waitAsync();
  __syncthreads();

  // ---- pipelined main loop (2 K-steps per trip) ----
  int kb = 0;
  while (true) {
    const int  k1 = kb + TILE_K;
    const bool m1 = (k1 < K);
    if (m1) { issueB(k1, bdst1); aR = cvt8_f32_to_f16(asrc + k1); }
    compute(0);
    if (!m1) break;
    *(v8h*)&As[1][arow][acol] = aR;
    waitAsync();
    __syncthreads();

    const int  k2 = kb + 2 * TILE_K;
    const bool m2 = (k2 < K);
    if (m2) { issueB(k2, bdst0); aR = cvt8_f32_to_f16(asrc + k2); }
    compute(1);
    if (!m2) break;
    *(v8h*)&As[0][arow][acol] = aR;
    waitAsync();
    __syncthreads();

    kb = k2;
  }

  // ---- store: lane<16 -> M=m0+r, N=lane;  lane>=16 -> M=m0+8+r, N=lane-16
  const int mOff = (lane >= 16) ? 8 : 0;
  #pragma unroll
  for (int i = 0; i < 2; ++i) {
    #pragma unroll
    for (int j = 0; j < 2; ++j) {
      const v8f accv = (i == 0) ? (j == 0 ? acc00 : acc01)
                                : (j == 0 ? acc10 : acc11);
      const int n = tileN + wn + j * 16 + (lane & 15);
      if (n < N) {
        const float bn = (epilogue == 1) ? bias[n] : 0.f;
        #pragma unroll
        for (int r = 0; r < 8; ++r) {
          const int m = tileM + wm + i * 16 + r + mOff;
          float v = accv[r];
          if (epilogue == 1) {
            v += bn;
            v = (v > 20.f) ? v : log1pf(expf(v));   // softplus
          }
          C[(size_t)m * ldc + n] = v;
        }
      }
    }
  }
}

// causal depthwise conv1d over xs = xz[:, :, 0:D_INNER], + bias, + SiLU
__global__ __launch_bounds__(256) void conv_silu_kernel(
    const float* __restrict__ xz, float* __restrict__ xc,
    const float* __restrict__ cw, const float* __restrict__ cb)
{
  const int idx = blockIdx.x * 256 + threadIdx.x;         // (t, d)
  const int t = idx / D_INNER;
  const int d = idx - t * D_INNER;
  const int b = t / SEQ;
  const int l = t - b * SEQ;

  float acc = cb[d];
  #pragma unroll
  for (int k = 0; k < D_CONV; ++k) {
    const int li = l + k - (D_CONV - 1);
    if (li >= 0)
      acc += xz[(size_t)(b * SEQ + li) * (2 * D_INNER) + d] * cw[d * D_CONV + k];
  }
  acc = acc / (1.f + expf(-acc));                         // SiLU
  xc[(size_t)t * D_INNER + d] = acc;
}

// selective scan: h[n] = h[n]*exp(delta*A[n]) + delta*u*B[n];  y = h.C + u*Dskip
// then y *= silu(z).  y overwrites xc in place.
__global__ __launch_bounds__(256) void scan_kernel(
    float* __restrict__ xc,            // u in, y out
    const float* __restrict__ delta,
    const float* __restrict__ xz,      // z = cols [D_INNER, 2*D_INNER)
    const float* __restrict__ dbl,     // B = cols [64,80), C = cols [80,96)
    const float* __restrict__ A_log,
    const float* __restrict__ D_skip)
{
  const int b = blockIdx.x >> 3;                          // 0..1
  const int d = (blockIdx.x & 7) * 256 + threadIdx.x;     // 0..2047

  float a[D_STATE], h[D_STATE];
  #pragma unroll
  for (int n = 0; n < D_STATE; ++n) {
    a[n] = -expf(A_log[(size_t)d * D_STATE + n]);
    h[n] = 0.f;
  }
  const float dsk = D_skip[d];

  __shared__ float sB[D_STATE];
  __shared__ float sC[D_STATE];

  for (int l = 0; l < SEQ; ++l) {
    const int t = b * SEQ + l;
    if (threadIdx.x < D_STATE)
      sB[threadIdx.x] = dbl[(size_t)t * 96 + DT_RANK + threadIdx.x];
    else if (threadIdx.x < 2 * D_STATE)
      sC[threadIdx.x - D_STATE] = dbl[(size_t)t * 96 + DT_RANK + D_STATE + (threadIdx.x - D_STATE)];
    __syncthreads();

    const float dlt = delta[(size_t)t * D_INNER + d];
    const float u   = xc[(size_t)t * D_INNER + d];
    const float du  = dlt * u;
    float y = 0.f;
    #pragma unroll
    for (int n = 0; n < D_STATE; ++n) {
      h[n] = h[n] * expf(dlt * a[n]) + du * sB[n];
      y += h[n] * sC[n];
    }
    y += u * dsk;
    const float z = xz[(size_t)t * (2 * D_INNER) + D_INNER + d];
    y *= z / (1.f + expf(-z));
    xc[(size_t)t * D_INNER + d] = y;
    __syncthreads();
  }
}

// out = LayerNorm(mm + xprev) ; one block per token, 4 channels/thread
__global__ __launch_bounds__(256) void resid_ln_kernel(
    const float* __restrict__ mm, const float* __restrict__ xprev,
    const float* __restrict__ lw, const float* __restrict__ lb,
    float* __restrict__ out)
{
  const int t = blockIdx.x;
  const int tid = threadIdx.x;
  __shared__ float red[256];

  float r[4];
  float s = 0.f;
  #pragma unroll
  for (int j = 0; j < 4; ++j) {
    const int c = tid + j * 256;
    r[j] = mm[(size_t)t * D_MODEL + c] + xprev[(size_t)t * D_MODEL + c];
    s += r[j];
  }
  red[tid] = s;
  __syncthreads();
  for (int off = 128; off > 0; off >>= 1) {
    if (tid < off) red[tid] += red[tid + off];
    __syncthreads();
  }
  const float mu = red[0] * (1.f / D_MODEL);
  __syncthreads();

  float v = 0.f;
  #pragma unroll
  for (int j = 0; j < 4; ++j) {
    const float dd = r[j] - mu;
    v += dd * dd;
  }
  red[tid] = v;
  __syncthreads();
  for (int off = 128; off > 0; off >>= 1) {
    if (tid < off) red[tid] += red[tid + off];
    __syncthreads();
  }
  const float rstd = rsqrtf(red[0] * (1.f / D_MODEL) + LN_EPS);

  #pragma unroll
  for (int j = 0; j < 4; ++j) {
    const int c = tid + j * 256;
    out[(size_t)t * D_MODEL + c] = (r[j] - mu) * rstd * lw[c] + lb[c];
  }
}

extern "C" void kernel_launch(void* const* d_in, const int* in_sizes, int n_in,
                              void* d_out, int out_size, void* d_ws, size_t ws_size,
                              hipStream_t stream) {
  (void)in_sizes; (void)n_in; (void)out_size; (void)ws_size;

  const float* x_in0     = (const float*)d_in[0];
  const float* in_proj_w = (const float*)d_in[1];
  const float* conv_w    = (const float*)d_in[2];
  const float* conv_b    = (const float*)d_in[3];
  const float* x_proj_w  = (const float*)d_in[4];
  const float* dt_proj_w = (const float*)d_in[5];
  const float* dt_proj_b = (const float*)d_in[6];
  const float* A_log     = (const float*)d_in[7];
  const float* D_skip    = (const float*)d_in[8];
  const float* out_proj_w= (const float*)d_in[9];
  const float* ln_w      = (const float*)d_in[10];
  const float* ln_b      = (const float*)d_in[11];
  float* out = (float*)d_out;

  // ---- workspace layout ----
  char* ws = (char*)d_ws;
  size_t off = 0;
  float* xz    = (float*)(ws + off); off += (size_t)TOKENS * 2 * D_INNER * 4;  // 32 MB
  float* xc    = (float*)(ws + off); off += (size_t)TOKENS * D_INNER * 4;      // 16 MB
  float* dbl   = (float*)(ws + off); off += (size_t)TOKENS * 96 * 4;           // 0.75 MB
  float* delta = (float*)(ws + off); off += (size_t)TOKENS * D_INNER * 4;      // 16 MB
  float* mmout = (float*)(ws + off); off += (size_t)TOKENS * D_MODEL * 4;      // 8 MB
  float* xbuf  = (float*)(ws + off); off += (size_t)TOKENS * D_MODEL * 4;      // 8 MB

  const size_t IPW_E = (size_t)4 * 2 * D_INNER * D_MODEL;  // 16.78M
  const size_t XPW_E = (size_t)4 * 96 * D_INNER;           // 0.79M
  const size_t DTW_E = (size_t)4 * D_INNER * DT_RANK;      // 0.52M
  const size_t OPW_E = (size_t)4 * D_MODEL * D_INNER;      // 8.39M
  _Float16* ipw_h = (_Float16*)(ws + off); off += IPW_E * 2;
  _Float16* xpw_h = (_Float16*)(ws + off); off += XPW_E * 2;
  _Float16* dtw_h = (_Float16*)(ws + off); off += DTW_E * 2;
  _Float16* opw_h = (_Float16*)(ws + off); off += OPW_E * 2;

  const dim3 blk(256);

  // ---- weight pre-conversion fp32 -> f16 (all layers at once) ----
  cvt_w_kernel<<<(int)(IPW_E / 8 / 256), blk, 0, stream>>>(in_proj_w,  ipw_h, (int)(IPW_E / 8));
  cvt_w_kernel<<<(int)(XPW_E / 8 / 256), blk, 0, stream>>>(x_proj_w,   xpw_h, (int)(XPW_E / 8));
  cvt_w_kernel<<<(int)(DTW_E / 8 / 256), blk, 0, stream>>>(dt_proj_w,  dtw_h, (int)(DTW_E / 8));
  cvt_w_kernel<<<(int)(OPW_E / 8 / 256), blk, 0, stream>>>(out_proj_w, opw_h, (int)(OPW_E / 8));

  for (int layer = 0; layer < 4; ++layer) {
    const float* xcur  = (layer == 0) ? x_in0 : xbuf;
    float*       lnout = (layer == 3) ? out   : xbuf;

    const _Float16* ipw = ipw_h + (size_t)layer * 2 * D_INNER * D_MODEL;
    const _Float16* xpw = xpw_h + (size_t)layer * 96 * D_INNER;
    const _Float16* dtw = dtw_h + (size_t)layer * D_INNER * DT_RANK;
    const _Float16* opw = opw_h + (size_t)layer * D_MODEL * D_INNER;
    const float* cw  = conv_w    + (size_t)layer * D_INNER * D_CONV;
    const float* cb  = conv_b    + (size_t)layer * D_INNER;
    const float* dtb = dt_proj_b + (size_t)layer * D_INNER;
    const float* al  = A_log     + (size_t)layer * D_INNER * D_STATE;
    const float* dsk = D_skip    + (size_t)layer * D_INNER;
    const float* lw  = ln_w      + (size_t)layer * D_MODEL;
    const float* lb  = ln_b      + (size_t)layer * D_MODEL;

    // 1) xz = xcur @ in_proj_w^T   [2048 x 4096], K=1024
    gemm_wmma_async_kernel<<<dim3((2 * D_INNER) / TILE_N, TOKENS / TILE_M), blk, 0, stream>>>(
        xcur, D_MODEL, ipw, D_MODEL, xz, 2 * D_INNER, 2 * D_INNER, D_MODEL, nullptr, 0);

    // 2) xc = silu(conv1d(xs) + cb)
    conv_silu_kernel<<<(TOKENS * D_INNER) / 256, blk, 0, stream>>>(xz, xc, cw, cb);

    // 3) dbl = xc @ x_proj_w^T     [2048 x 96], K=2048
    gemm_wmma_async_kernel<<<dim3((96 + TILE_N - 1) / TILE_N, TOKENS / TILE_M), blk, 0, stream>>>(
        xc, D_INNER, xpw, D_INNER, dbl, 96, 96, D_INNER, nullptr, 0);

    // 4) delta = softplus(dbl[:, :64] @ dt_proj_w^T + dtb)  [2048 x 2048], K=64
    gemm_wmma_async_kernel<<<dim3(D_INNER / TILE_N, TOKENS / TILE_M), blk, 0, stream>>>(
        dbl, 96, dtw, DT_RANK, delta, D_INNER, D_INNER, DT_RANK, dtb, 1);

    // 5) selective scan (y written over xc), fused * silu(z)
    scan_kernel<<<BATCH * (D_INNER / 256), blk, 0, stream>>>(xc, delta, xz, dbl, al, dsk);

    // 6) mmout = y @ out_proj_w^T  [2048 x 1024], K=2048
    gemm_wmma_async_kernel<<<dim3(D_MODEL / TILE_N, TOKENS / TILE_M), blk, 0, stream>>>(
        xc, D_INNER, opw, D_INNER, mmout, D_MODEL, D_MODEL, D_INNER, nullptr, 0);

    // 7) x_next = LayerNorm(mmout + xcur)
    resid_ln_kernel<<<TOKENS, blk, 0, stream>>>(mmout, xcur, lw, lb, lnout);
  }
}
